// WindowAttentionGlobal_30124900614407
// MI455X (gfx1250) — matrix operand order repaired and checked by reading, not compiled, not run on hardware
//
#include <hip/hip_runtime.h>
#include <hip/hip_bf16.h>
#include <cstdint>
#include <cstddef>

#define NTOK 49
#define NPAD 64
#define CDIM 128
#define NH   4
#define HD   32
#define NIMG 16
#define WPB  8
#define TPB  256

typedef __attribute__((ext_vector_type(16))) __bf16 v16bf;
typedef __attribute__((ext_vector_type(8)))  float  v8f;
typedef __attribute__((ext_vector_type(4)))  unsigned int v4u;
typedef __attribute__((ext_vector_type(8)))  int v8i;
typedef __attribute__((ext_vector_type(4)))  int v4i;

// ---- LDS arena layout (bytes) ----
#define OFF_WKV  0        // bf16 [256][128]  qkv weight (persistent)
#define OFF_WPJ  65536    // bf16 [128][128]  proj weight (persistent)
#define OFF_X    98304    // bf16 [64][128]   x window (reused as AO)
#define OFF_Q    114688   // bf16 [4][64][32] q (pre-scaled)
#define OFF_K    131072   // bf16 [4][64][32] k, [h][n][d]
#define OFF_VT   147456   // bf16 [4][32][64] v transposed, [h][d][n]
#define OFF_S    163840   // f32  [64][64]    logits (per head)
#define OFF_P    180224   // bf16 [64][64]    probs (per head)
#define OFF_BIAS 188416   // bf16 [4][64][64] masked rel-pos bias (persistent)
#define OFF_RED  221184   // f32  [256]       softmax max reduction
#define OFF_RED2 222208   // f32  [256]       softmax sum reduction
#define OFF_OS   223232   // f32  [64][128]   output staging
#define OFF_XST  256000   // f32  [49*128]    TDM staging for next window's x
#define OFF_BQ   281088   // f32  [256]       qkv bias (persistent)
#define OFF_BP   282112   // f32  [128]       proj bias (persistent)
#define SMEM_BYTES 282624

#define XWIN (NTOK * CDIM)   // 6272 floats per window

__device__ __forceinline__ __bf16 f2bf(float f) {
  unsigned int u = __builtin_bit_cast(unsigned int, f);
  u += 0x7FFFu + ((u >> 16) & 1u);              // round-to-nearest-even
  unsigned short s = (unsigned short)(u >> 16);
  return __builtin_bit_cast(__bf16, s);
}
__device__ __forceinline__ float bf2f(__bf16 b) {
  unsigned int u = ((unsigned int)__builtin_bit_cast(unsigned short, b)) << 16;
  return __builtin_bit_cast(float, u);
}
__device__ __forceinline__ unsigned int pack2(float a, float b) {
  return (unsigned int)__builtin_bit_cast(unsigned short, f2bf(a)) |
         ((unsigned int)__builtin_bit_cast(unsigned short, f2bf(b)) << 16);
}

// Tensor Data Mover: async DMA one window of x (6272 f32, contiguous) to LDS.
// D# built per CDNA5 ISA ch.8: group0 = {ctrl, lds_addr, gaddr_lo, gaddr_hi|type},
// group1 = dims/strides/tile (data_size=4B, tensor_dim0=tile_dim0=6272).
__device__ __forceinline__ void tdm_load_x(const float* gptr, unsigned lds_addr) {
  unsigned long long ga = (unsigned long long)(const void*)gptr;
  v4u g0;
  g0[0] = 1u;                                        // count=1, user descriptor
  g0[1] = lds_addr;                                  // LDS byte address
  g0[2] = (unsigned)(ga & 0xFFFFFFFFu);              // global_addr[31:0]
  g0[3] = (unsigned)((ga >> 32) & 0x01FFFFFFu) | (2u << 30);  // [56:32] | type=2
  v8i g1;
  g1[0] = (int)(2u << 16);           // data_size=2 (4 bytes); no multicast/pad
  g1[1] = (int)((XWIN & 0xFFFF) << 16);  // tensor_dim0[15:0] @ bits[63:48]
  g1[2] = (int)((1u << 16) | ((XWIN >> 16) & 0xFFFF)); // dim0 hi | tensor_dim1=1
  g1[3] = (int)((unsigned)XWIN << 16);   // tile_dim0 @ bits[127:112]
  g1[4] = 1;                             // tile_dim1 = 1 (single row)
  g1[5] = XWIN;                          // tensor_dim0_stride[31:0]
  g1[6] = 0;
  g1[7] = 0;
  v4i gz4 = {0, 0, 0, 0};
  v8i gz8 = {0, 0, 0, 0, 0, 0, 0, 0};
  __builtin_amdgcn_tensor_load_to_lds(g0, g1, gz4, gz4, gz8, 0);
}

// A fragment: 16x32 bf16 tile from row-major [M][K] storage (ld = K stride, elems).
__device__ __forceinline__ v16bf load_a_frag(const __bf16* base, int ld,
                                             int m0, int k0, int lane) {
  union { unsigned int u[8]; v16bf v; } f;
  const int half = lane >> 4;
  const __bf16* row = base + (m0 + (lane & 15)) * ld;
#pragma unroll
  for (int j = 0; j < 8; ++j) {
    int kb = k0 + 8 * half + ((j < 4) ? 2 * j : 16 + 2 * (j - 4));
    f.u[j] = *(const unsigned int*)(row + kb);
  }
  return f.v;
}
// B fragment: 32x16 bf16 tile, storage [N][K] row-major (ld = K stride).
__device__ __forceinline__ v16bf load_b_frag_nk(const __bf16* base, int ld,
                                                int n0, int k0, int lane) {
  union { unsigned int u[8]; v16bf v; } f;
  const int half = lane >> 4;
  const __bf16* row = base + (n0 + (lane & 15)) * ld;
#pragma unroll
  for (int j = 0; j < 8; ++j) {
    int kb = k0 + 8 * half + ((j < 4) ? 2 * j : 16 + 2 * (j - 4));
    f.u[j] = *(const unsigned int*)(row + kb);
  }
  return f.v;
}
#define WMMA_BF16(A, B, C) __builtin_amdgcn_wmma_f32_16x16x32_bf16( \
    false, (A), false, (B), (short)0, (C), false, false)

__global__ void winattn_kernel(const float* __restrict__ x,
                               const float* __restrict__ qg,
                               const float* __restrict__ qkvw,
                               const float* __restrict__ qkvb,
                               const float* __restrict__ pw,
                               const float* __restrict__ pb,
                               const float* __restrict__ rel,
                               float* __restrict__ out,
                               int nwin) {
  extern __shared__ char smem[];
  __bf16* wkv = (__bf16*)(smem + OFF_WKV);
  __bf16* wpj = (__bf16*)(smem + OFF_WPJ);
  __bf16* xs  = (__bf16*)(smem + OFF_X);    // also AO
  __bf16* qs  = (__bf16*)(smem + OFF_Q);
  __bf16* ks  = (__bf16*)(smem + OFF_K);
  __bf16* vts = (__bf16*)(smem + OFF_VT);
  float*  ss  = (float*) (smem + OFF_S);
  __bf16* ps  = (__bf16*)(smem + OFF_P);
  __bf16* bs  = (__bf16*)(smem + OFF_BIAS);
  float*  red = (float*) (smem + OFF_RED);
  float*  red2= (float*) (smem + OFF_RED2);
  float*  os  = (float*) (smem + OFF_OS);
  float*  xst = (float*) (smem + OFF_XST);
  float*  bq  = (float*) (smem + OFF_BQ);
  float*  bp  = (float*) (smem + OFF_BP);
  const unsigned lds_xst = (unsigned)(uintptr_t)(smem + OFF_XST);

  const int tid  = threadIdx.x;
  const int lane = tid & 31;
  const int wave = tid >> 5;
  const int half = lane >> 4;
  const float scale = 0.17677669529663687f;   // 32^-0.5

  // kick off async TDM load of the first window's x while we stage weights
  const int win0 = blockIdx.x * WPB;
  if (wave == 0 && win0 < nwin)
    tdm_load_x(x + (size_t)win0 * XWIN, lds_xst);

  // ---- per-block persistent staging: weights (bf16), biases, rel-pos bias
  for (int i4 = tid; i4 < (256 * CDIM) / 4; i4 += TPB) {
    float4 v = *(const float4*)(qkvw + i4 * 4);
    *(uint2*)(wkv + i4 * 4) = make_uint2(pack2(v.x, v.y), pack2(v.z, v.w));
  }
  for (int i4 = tid; i4 < (CDIM * CDIM) / 4; i4 += TPB) {
    float4 v = *(const float4*)(pw + i4 * 4);
    *(uint2*)(wpj + i4 * 4) = make_uint2(pack2(v.x, v.y), pack2(v.z, v.w));
  }
  if (tid < 256) bq[tid] = qkvb[tid];
  if (tid < 128) bp[tid] = pb[tid];
  for (int i = tid; i < NH * NPAD * NPAD; i += TPB) {
    int h = i >> 12, m = (i >> 6) & 63, nc = i & 63;
    float b;
    if (nc >= NTOK)      b = -1e30f;   // key-padding mask folded into bias
    else if (m >= NTOK)  b = 0.f;
    else {
      int rq = m / 7, cq = m - 7 * rq, rk = nc / 7, ck = nc - 7 * rk;
      b = rel[((rq - rk + 6) * 13 + (cq - ck + 6)) * NH + h];
    }
    bs[i] = f2bf(b);
  }
  __syncthreads();

  for (int wloc = 0; wloc < WPB; ++wloc) {
    const int win = blockIdx.x * WPB + wloc;
    if (win >= nwin) break;                    // uniform per block
    const int img = win & (NIMG - 1);

    // wait for the TDM transfer of this window's x, then publish to all waves
    if (wave == 0) __builtin_amdgcn_s_wait_tensorcnt(0);
    __syncthreads();

    // ---- convert staged x (LDS f32) -> bf16 xs (padded); stage pre-scaled q
    for (int i4 = tid; i4 < (NPAD * CDIM) / 4; i4 += TPB) {
      int n = i4 >> 5, c = (i4 & 31) * 4;
      float4 v = make_float4(0.f, 0.f, 0.f, 0.f);
      if (n < NTOK) v = *(const float4*)(xst + n * CDIM + c);
      *(uint2*)(xs + i4 * 4) = make_uint2(pack2(v.x, v.y), pack2(v.z, v.w));
    }
    for (int i4 = tid; i4 < (NH * NPAD * HD) / 4; i4 += TPB) {
      int h = i4 >> 9, n = (i4 >> 3) & 63, d = (i4 & 7) * 4;
      float4 v = make_float4(0.f, 0.f, 0.f, 0.f);
      if (n < NTOK) {
        v = *(const float4*)(qg + (((size_t)img * NTOK + n) * NH + h) * HD + d);
        v.x *= scale; v.y *= scale; v.z *= scale; v.w *= scale;
      }
      *(uint2*)(qs + i4 * 4) = make_uint2(pack2(v.x, v.y), pack2(v.z, v.w));
    }
    __syncthreads();   // xs/qs ready; xst is now free

    // async: start fetching next window's x while we compute this one
    if (wave == 0 && wloc + 1 < WPB && win + 1 < nwin)
      tdm_load_x(x + (size_t)(win + 1) * XWIN, lds_xst);

    const int mt = wave & 3, ntb = wave >> 2;  // per-wave tile coordinates

    // ---- Stage A (K half): dual-accumulator, shared A fragments
#pragma unroll
    for (int pp = 0; pp < 2; ++pp) {
      int nt0 = ntb + pp * 4, nt1 = nt0 + 2;   // nt in 0..7 -> j < 128
      v8f acc0 = {}, acc1 = {};
#pragma unroll
      for (int kk = 0; kk < 4; ++kk) {
        v16bf a  = load_a_frag(xs, CDIM, mt * 16, kk * 32, lane);
        v16bf b0 = load_b_frag_nk(wkv, CDIM, nt0 * 16, kk * 32, lane);
        v16bf b1 = load_b_frag_nk(wkv, CDIM, nt1 * 16, kk * 32, lane);
        acc0 = WMMA_BF16(a, b0, acc0);
        acc1 = WMMA_BF16(a, b1, acc1);
      }
#pragma unroll
      for (int s = 0; s < 2; ++s) {
        v8f acc = s ? acc1 : acc0;
        int j = (s ? nt1 : nt0) * 16 + (lane & 15);
        float bias = bq[j];
        int hh = j >> 5, d = j & 31;
#pragma unroll
        for (int r = 0; r < 8; ++r)
          ks[(hh * NPAD + mt * 16 + r + 8 * half) * HD + d] = f2bf(acc[r] + bias);
      }
    }
    // ---- Stage A (V half): -> vts [h][d][n] transposed, pair-packed stores
#pragma unroll
    for (int pp = 0; pp < 2; ++pp) {
      int nt0 = 8 + ntb + pp * 4, nt1 = nt0 + 2;   // nt 8..15 -> j >= 128
      v8f acc0 = {}, acc1 = {};
#pragma unroll
      for (int kk = 0; kk < 4; ++kk) {
        v16bf a  = load_a_frag(xs, CDIM, mt * 16, kk * 32, lane);
        v16bf b0 = load_b_frag_nk(wkv, CDIM, nt0 * 16, kk * 32, lane);
        v16bf b1 = load_b_frag_nk(wkv, CDIM, nt1 * 16, kk * 32, lane);
        acc0 = WMMA_BF16(a, b0, acc0);
        acc1 = WMMA_BF16(a, b1, acc1);
      }
#pragma unroll
      for (int s = 0; s < 2; ++s) {
        v8f acc = s ? acc1 : acc0;
        int j = (s ? nt1 : nt0) * 16 + (lane & 15);
        float bias = bq[j];
        int hh = (j >> 5) & 3, d = j & 31;
        __bf16* vrow = vts + (hh * HD + d) * NPAD;
#pragma unroll
        for (int r = 0; r < 8; r += 2) {
          int m = mt * 16 + r + 8 * half;       // even -> 4B aligned
          *(unsigned int*)(vrow + m) = pack2(acc[r] + bias, acc[r + 1] + bias);
        }
      }
    }
    __syncthreads();

    // ---- per-head attention
    for (int h = 0; h < NH; ++h) {
      const __bf16* qh  = qs + h * NPAD * HD;
      const __bf16* kh  = ks + h * NPAD * HD;
      const __bf16* vth = vts + h * HD * NPAD;
      const __bf16* bh  = bs + h * NPAD * NPAD;

      // Stage B: S = q @ k^T + bias (mask pre-folded); dual tiles per wave
      {
        int nt0 = ntb, nt1 = ntb + 2;
        v16bf a  = load_a_frag(qh, HD, mt * 16, 0, lane);
        v16bf b0 = load_b_frag_nk(kh, HD, nt0 * 16, 0, lane);
        v16bf b1 = load_b_frag_nk(kh, HD, nt1 * 16, 0, lane);
        v8f acc0 = {}, acc1 = {};
        acc0 = WMMA_BF16(a, b0, acc0);
        acc1 = WMMA_BF16(a, b1, acc1);
#pragma unroll
        for (int s = 0; s < 2; ++s) {
          v8f acc = s ? acc1 : acc0;
          int nc = (s ? nt1 : nt0) * 16 + (lane & 15);
#pragma unroll
          for (int r = 0; r < 8; ++r) {
            int m = mt * 16 + r + 8 * half;
            ss[m * NPAD + nc] = acc[r] + bf2f(bh[m * NPAD + nc]);
          }
        }
      }
      __syncthreads();

      // softmax: 4 threads per row (64 rows x 16-col quarters)
      {
        int row = tid >> 2, qt = tid & 3, c0 = qt * 16;
        float* srow = ss + row * NPAD;
        float lmax = -3e38f;
#pragma unroll
        for (int c = 0; c < 16; ++c) lmax = fmaxf(lmax, srow[c0 + c]);
        red[tid] = lmax;
        __syncthreads();
        float mx = fmaxf(fmaxf(red[row * 4 + 0], red[row * 4 + 1]),
                         fmaxf(red[row * 4 + 2], red[row * 4 + 3]));
        float lsum = 0.f;
#pragma unroll
        for (int c = 0; c < 16; ++c) {
          float e = __expf(srow[c0 + c] - mx);   // masked cols -> 0
          srow[c0 + c] = e;
          lsum += e;
        }
        red2[tid] = lsum;
        __syncthreads();
        float sum = red2[row * 4 + 0] + red2[row * 4 + 1] +
                    red2[row * 4 + 2] + red2[row * 4 + 3];
        float rinv = 1.f / sum;
#pragma unroll
        for (int c = 0; c < 16; c += 2)
          *(unsigned int*)(ps + row * NPAD + c0 + c) =
              pack2(srow[c0 + c] * rinv, srow[c0 + c + 1] * rinv);
      }
      __syncthreads();

      // Stage C: AO_h[64x32] = P @ V_h ; split K-chain into 2 independent wmmas
      {
        int dt = wave >> 2;                      // 8 tiles, one per wave
        v16bf a0 = load_a_frag(ps, NPAD, mt * 16, 0, lane);
        v16bf b0 = load_b_frag_nk(vth, NPAD, dt * 16, 0, lane);
        v16bf a1 = load_a_frag(ps, NPAD, mt * 16, 32, lane);
        v16bf b1 = load_b_frag_nk(vth, NPAD, dt * 16, 32, lane);
        v8f acc0 = {}, acc1 = {};
        acc0 = WMMA_BF16(a0, b0, acc0);
        acc1 = WMMA_BF16(a1, b1, acc1);
        int d = dt * 16 + (lane & 15);
#pragma unroll
        for (int r = 0; r < 8; ++r) {
          int m = mt * 16 + r + 8 * half;
          xs[m * CDIM + h * HD + d] = f2bf(acc0[r] + acc1[r]);  // AO in x region
        }
      }
      __syncthreads();
    }

    // ---- Stage D: OUT = AO @ Wproj^T + b; dual tiles -> LDS staging
#pragma unroll
    for (int pp = 0; pp < 2; ++pp) {
      int nt0 = ntb + pp * 4, nt1 = nt0 + 2;
      v8f acc0 = {}, acc1 = {};
#pragma unroll
      for (int kk = 0; kk < 4; ++kk) {
        v16bf a  = load_a_frag(xs, CDIM, mt * 16, kk * 32, lane);
        v16bf b0 = load_b_frag_nk(wpj, CDIM, nt0 * 16, kk * 32, lane);
        v16bf b1 = load_b_frag_nk(wpj, CDIM, nt1 * 16, kk * 32, lane);
        acc0 = WMMA_BF16(a, b0, acc0);
        acc1 = WMMA_BF16(a, b1, acc1);
      }
#pragma unroll
      for (int s = 0; s < 2; ++s) {
        v8f acc = s ? acc1 : acc0;
        int j = (s ? nt1 : nt0) * 16 + (lane & 15);
        float bias = bp[j];
#pragma unroll
        for (int r = 0; r < 8; ++r)
          os[(mt * 16 + r + 8 * half) * CDIM + j] = acc[r] + bias;
      }
    }
    __syncthreads();
    {
      float* owin = out + (size_t)win * XWIN;
      for (int i4 = tid; i4 < XWIN / 4; i4 += TPB)
        *(float4*)(owin + i4 * 4) = *(const float4*)(os + i4 * 4);
    }
    __syncthreads();   // protect AO/q/k/v/os before next window's staging
  }
}

extern "C" void kernel_launch(void* const* d_in, const int* in_sizes, int n_in,
                              void* d_out, int out_size, void* d_ws, size_t ws_size,
                              hipStream_t stream) {
  const float* x    = (const float*)d_in[0];
  const float* qg   = (const float*)d_in[1];
  const float* qkvw = (const float*)d_in[2];
  const float* qkvb = (const float*)d_in[3];
  const float* pw   = (const float*)d_in[4];
  const float* pb   = (const float*)d_in[5];
  const float* rel  = (const float*)d_in[6];
  float* out = (float*)d_out;

  int nwin   = in_sizes[0] / XWIN;                // 16384
  int blocks = (nwin + WPB - 1) / WPB;            // 2048

  winattn_kernel<<<blocks, TPB, SMEM_BYTES, stream>>>(
      x, qg, qkvw, qkvb, pw, pb, rel, out, nwin);
}
